// MultiLayerRNN_59966333386916
// MI455X (gfx1250) — matrix operand order repaired and checked by reading, not compiled, not run on hardware
//
#include <hip/hip_runtime.h>
#include <hip/hip_bf16.h>

typedef __attribute__((ext_vector_type(16))) _Float16 v16h;
typedef __attribute__((ext_vector_type(8)))  _Float16 v8h;
typedef __attribute__((ext_vector_type(8)))  float    v8f;

#define RNN_B   64
#define RNN_S   1024
#define RNN_E   256
#define RNN_H   512
#define LDS_LDW 520   // padded row stride (halves) to avoid bank conflicts

// ---------------------------------------------------------------------------
// Fragment loaders (wave32, 16x16x32 f16 WMMA layouts per CDNA5 ISA 7.12.2)
// A (MxK, row-major, ld in halves): lane m=l&15, sel=l>>4
//   elems 0..7  = K[k0+sel*8   .. +7], elems 8..15 = K[k0+16+sel*8 .. +7]
// ---------------------------------------------------------------------------
__device__ __forceinline__ v16h load_a16(const _Float16* base, int ld, int k0) {
  const int l = threadIdx.x & 31, m = l & 15, sel = l >> 4;
  const _Float16* p = base + m * ld + k0 + sel * 8;
  v8h lo = *(const v8h*)(p);
  v8h hi = *(const v8h*)(p + 16);
  v16h r;
#pragma unroll
  for (int i = 0; i < 8; ++i) { r[i] = lo[i]; r[i + 8] = hi[i]; }
  return r;
}

// B (KxN) fed from transposed storage BT (NxK, row-major, ld in halves):
// lane n=l&15, sel=l>>4; elems 0..7 = K[k0+sel*8 .. +7], elems 8..15 = K[k0+16+sel*8 .. +7]
__device__ __forceinline__ v16h load_b16(const _Float16* baseT, int ld, int k0) {
  const int l = threadIdx.x & 31, n = l & 15, sel = l >> 4;
  const _Float16* p = baseT + n * ld + k0 + sel * 8;
  v8h lo = *(const v8h*)(p);
  v8h hi = *(const v8h*)(p + 16);
  v16h r;
#pragma unroll
  for (int i = 0; i < 8; ++i) { r[i] = lo[i]; r[i + 8] = hi[i]; }
  return r;
}

// ---------------------------------------------------------------------------
// Prep: transpose f32 [R][C] -> f16 [C][R]
// ---------------------------------------------------------------------------
__global__ void k_transpose_f16(const float* __restrict__ in,
                                _Float16* __restrict__ out, int R, int C) {
  int i = blockIdx.x * blockDim.x + threadIdx.x;
  if (i >= R * C) return;
  int c = i / R, r = i % R;                  // writes contiguous in r
  out[(size_t)c * R + r] = (_Float16)in[(size_t)r * C + c];
}

// ---------------------------------------------------------------------------
// pre0[t*64+b][n] = emb[x[b][t]] @ Wxh0 + bxh0 + bhh0   (f16 out, f32 accum)
// One 16x16 output tile per wave; embedding gather fused into A-fragment load.
// ---------------------------------------------------------------------------
__global__ void k_pre0(const int* __restrict__ x, const float* __restrict__ emb,
                       const _Float16* __restrict__ WxT0,
                       const float* __restrict__ bxh0,
                       const float* __restrict__ bhh0,
                       _Float16* __restrict__ pre0) {
  const int w    = (blockIdx.x * blockDim.x + threadIdx.x) >> 5;
  const int lane = threadIdx.x & 31, m = lane & 15, sel = lane >> 4;
  const int tr = w >> 5;            // 0..4095 row tiles over S*B
  const int tc = w & 31;            // 0..31  col tiles over HID

  const int row = tr * 16 + m;      // row = t*B + b
  const int t = row >> 6, b = row & 63;
  const int tok = x[b * RNN_S + t];
  const float* arow = emb + (size_t)tok * RNN_E;

  v8f c = {};
#pragma unroll
  for (int kk = 0; kk < RNN_E / 32; ++kk) {
    const int k0 = kk * 32;
    v16h a;
    const float* pa0 = arow + k0 + sel * 8;
    const float* pa1 = arow + k0 + 16 + sel * 8;
#pragma unroll
    for (int i = 0; i < 8; ++i) { a[i] = (_Float16)pa0[i]; a[i + 8] = (_Float16)pa1[i]; }
    v16h bf = load_b16(WxT0 + (size_t)tc * 16 * RNN_E, RNN_E, k0);
    c = __builtin_amdgcn_wmma_f32_16x16x32_f16(false, a, false, bf, (short)0, c,
                                               false, false);
  }
  const int col  = tc * 16 + m;
  const float bi = bxh0[col] + bhh0[col];
#pragma unroll
  for (int j = 0; j < 8; ++j) {
    const int orow = tr * 16 + j + 8 * sel;
    pre0[(size_t)orow * RNN_H + col] = (_Float16)(c[j] + bi);
  }
}

// ---------------------------------------------------------------------------
// Persistent recurrence: one workgroup (32 waves), h0/h1 resident in LDS.
// Per step: h0 = tanh(pre0[t] + h0@Whh0); h1 = tanh(b1 + h0@Wxh1 + h1@Whh1)
// Each wave owns 4 of the 128 16x16 output tiles (fixed row-tile, 4 col-tiles).
// Tail: out[b] = sigmoid(h1[b] . fc_w + fc_b)
// ---------------------------------------------------------------------------
__global__ void __launch_bounds__(1024)
k_rnn(const _Float16* __restrict__ pre0,
      const _Float16* __restrict__ WhT0,   // [H][H]  Whh0^T
      const _Float16* __restrict__ WxT1,   // [H][H]  Wxh1^T
      const _Float16* __restrict__ WhT1,   // [H][H]  Whh1^T
      const float* __restrict__ bxh1, const float* __restrict__ bhh1,
      const float* __restrict__ fc_w, const float* __restrict__ fc_b,
      float* __restrict__ out) {
  __shared__ _Float16 sh0[RNN_B][LDS_LDW];
  __shared__ _Float16 sh1[RNN_B][LDS_LDW];

  // zero hidden state
  {
    _Float16* z0 = &sh0[0][0];
    _Float16* z1 = &sh1[0][0];
    for (int i = threadIdx.x; i < RNN_B * LDS_LDW; i += 1024) {
      z0[i] = (_Float16)0.f; z1[i] = (_Float16)0.f;
    }
  }
  __syncthreads();

  const int wave = threadIdx.x >> 5;
  const int lane = threadIdx.x & 31, m = lane & 15, sel = lane >> 4;
  const int tm  = wave & 3;          // row tile (0..3) -> rows tm*16..+15
  const int tnb = (wave >> 2) * 4;   // first of 4 consecutive col tiles

  float bias1[4];
#pragma unroll
  for (int i = 0; i < 4; ++i) {
    const int col = (tnb + i) * 16 + m;
    bias1[i] = bxh1[col] + bhh1[col];
  }

  for (int t = 0; t < RNN_S; ++t) {
    if (t + 1 < RNN_S) {  // best-effort prefetch of next step's pre-activations
      __builtin_prefetch(pre0 + (size_t)(t + 1) * RNN_B * RNN_H +
                             (size_t)(tm * 16 + m) * RNN_H + tnb * 16, 0, 0);
    }

    // ---- layer 0: c = pre0[t] + h0_old @ Whh0 ----
    v8f c[4];
#pragma unroll
    for (int i = 0; i < 4; ++i) {
      const int col = (tnb + i) * 16 + m;
#pragma unroll
      for (int j = 0; j < 8; ++j) {
        const int row = t * RNN_B + tm * 16 + j + 8 * sel;
        c[i][j] = (float)pre0[(size_t)row * RNN_H + col];
      }
    }
    for (int kk = 0; kk < RNN_H / 32; ++kk) {
      const int k0 = kk * 32;
      v16h a = load_a16(&sh0[tm * 16][0], LDS_LDW, k0);
#pragma unroll
      for (int i = 0; i < 4; ++i) {
        v16h bf = load_b16(WhT0 + (size_t)(tnb + i) * 16 * RNN_H, RNN_H, k0);
        c[i] = __builtin_amdgcn_wmma_f32_16x16x32_f16(false, a, false, bf,
                                                      (short)0, c[i], false, false);
      }
    }
    __syncthreads();                       // all h0_old reads done
#pragma unroll
    for (int i = 0; i < 4; ++i) {
      const int col = (tnb + i) * 16 + m;
#pragma unroll
      for (int j = 0; j < 8; ++j)
        sh0[tm * 16 + j + 8 * sel][col] = (_Float16)tanhf(c[i][j]);
    }
    __syncthreads();                       // h0_new visible

    // ---- layer 1: c = bias1 + h0_new @ Wxh1 + h1_old @ Whh1 ----
#pragma unroll
    for (int i = 0; i < 4; ++i)
#pragma unroll
      for (int j = 0; j < 8; ++j) c[i][j] = bias1[i];

    for (int kk = 0; kk < RNN_H / 32; ++kk) {
      const int k0 = kk * 32;
      v16h a = load_a16(&sh0[tm * 16][0], LDS_LDW, k0);
#pragma unroll
      for (int i = 0; i < 4; ++i) {
        v16h bf = load_b16(WxT1 + (size_t)(tnb + i) * 16 * RNN_H, RNN_H, k0);
        c[i] = __builtin_amdgcn_wmma_f32_16x16x32_f16(false, a, false, bf,
                                                      (short)0, c[i], false, false);
      }
    }
    for (int kk = 0; kk < RNN_H / 32; ++kk) {
      const int k0 = kk * 32;
      v16h a = load_a16(&sh1[tm * 16][0], LDS_LDW, k0);
#pragma unroll
      for (int i = 0; i < 4; ++i) {
        v16h bf = load_b16(WhT1 + (size_t)(tnb + i) * 16 * RNN_H, RNN_H, k0);
        c[i] = __builtin_amdgcn_wmma_f32_16x16x32_f16(false, a, false, bf,
                                                      (short)0, c[i], false, false);
      }
    }
    __syncthreads();                       // all h1_old reads done
#pragma unroll
    for (int i = 0; i < 4; ++i) {
      const int col = (tnb + i) * 16 + m;
#pragma unroll
      for (int j = 0; j < 8; ++j)
        sh1[tm * 16 + j + 8 * sel][col] = (_Float16)tanhf(c[i][j]);
    }
    __syncthreads();                       // h1_new visible
  }

  // ---- head: sigmoid(h1 @ fc_w + fc_b) ----
  if (threadIdx.x < RNN_B) {
    const int b = threadIdx.x;
    float s = fc_b[0];
    for (int n = 0; n < RNN_H; ++n) s += (float)sh1[b][n] * fc_w[n];
    out[b] = 1.f / (1.f + __expf(-s));
  }
}

// ---------------------------------------------------------------------------
extern "C" void kernel_launch(void* const* d_in, const int* in_sizes, int n_in,
                              void* d_out, int out_size, void* d_ws, size_t ws_size,
                              hipStream_t stream) {
  const int*   x    = (const int*)d_in[0];
  const float* emb  = (const float*)d_in[1];
  const float* Wxh0 = (const float*)d_in[2];
  const float* bxh0 = (const float*)d_in[3];
  const float* Whh0 = (const float*)d_in[4];
  const float* bhh0 = (const float*)d_in[5];
  const float* Wxh1 = (const float*)d_in[6];
  const float* bxh1 = (const float*)d_in[7];
  const float* Whh1 = (const float*)d_in[8];
  const float* bhh1 = (const float*)d_in[9];
  const float* fc_w = (const float*)d_in[10];
  const float* fc_b = (const float*)d_in[11];
  float* out = (float*)d_out;

  // workspace layout (f16)
  char* p = (char*)d_ws;
  _Float16* WxT0 = (_Float16*)p;                          p += (size_t)RNN_H * RNN_E * 2;
  _Float16* WhT0 = (_Float16*)p;                          p += (size_t)RNN_H * RNN_H * 2;
  _Float16* WxT1 = (_Float16*)p;                          p += (size_t)RNN_H * RNN_H * 2;
  _Float16* WhT1 = (_Float16*)p;                          p += (size_t)RNN_H * RNN_H * 2;
  _Float16* pre0 = (_Float16*)p;  // [S*B][H] f16

  // 1) transpose/convert weights to f16 [N][K]
  {
    int n0 = RNN_E * RNN_H, n1 = RNN_H * RNN_H;
    k_transpose_f16<<<(n0 + 255) / 256, 256, 0, stream>>>(Wxh0, WxT0, RNN_E, RNN_H);
    k_transpose_f16<<<(n1 + 255) / 256, 256, 0, stream>>>(Whh0, WhT0, RNN_H, RNN_H);
    k_transpose_f16<<<(n1 + 255) / 256, 256, 0, stream>>>(Wxh1, WxT1, RNN_H, RNN_H);
    k_transpose_f16<<<(n1 + 255) / 256, 256, 0, stream>>>(Whh1, WhT1, RNN_H, RNN_H);
  }

  // 2) fused embedding-gather + input GEMM (whole-GPU WMMA)
  {
    const int tiles  = (RNN_S * RNN_B / 16) * (RNN_H / 16);  // 131072
    const int blocks = tiles / 8;                            // 8 waves/block
    k_pre0<<<blocks, 256, 0, stream>>>(x, emb, WxT0, bxh0, bhh0, pre0);
  }

  // 3) persistent single-WGP recurrence + head
  k_rnn<<<1, 1024, 0, stream>>>(pre0, WhT0, WxT1, WhT1, bxh1, bhh1, fc_w, fc_b, out);
}
